// PerClassGating_40630390620873
// MI455X (gfx1250) — matrix operand order repaired and checked by reading, not compiled, not run on hardware
//
#include <hip/hip_runtime.h>
#include <hip/hip_bf16.h>

typedef __attribute__((ext_vector_type(16))) __bf16 v16bf;
typedef __attribute__((ext_vector_type(8)))  float   v8f;

__device__ __forceinline__ v8f wmma_bf16(v16bf a, v16bf b, v8f c) {
  // 8 args: (neg_a, A, neg_b, B, c_mod, C, reuse_a, reuse_b)
  return __builtin_amdgcn_wmma_f32_16x16x32_bf16(false, a, false, b, (short)0, c,
                                                 false, false);
}

__device__ __forceinline__ v8f vzero8() {
  v8f z;
#pragma unroll
  for (int i = 0; i < 8; ++i) z[i] = 0.0f;
  return z;
}

// A fragment (16x32 bf16) built from fp32 row-major [M x K], lda = K.
// Layout: lanes 0-15 row m0+l, K = {k0..k0+7, k0+16..k0+23};
//         lanes 16-31 row m0+l-16, K shifted by +8.
__device__ __forceinline__ v16bf load_a_f32(const float* __restrict__ A, int lda,
                                            int m0, int k0, int lane) {
  const int row = m0 + (lane & 15);
  const int kb  = k0 + ((lane >> 4) << 3);
  const float* p = A + (size_t)row * lda + kb;
  v16bf r;
#pragma unroll
  for (int i = 0; i < 8; ++i) r[i] = (__bf16)p[i];
#pragma unroll
  for (int i = 0; i < 8; ++i) r[8 + i] = (__bf16)p[16 + i];
  return r;
}

// A fragment (16x32 bf16) from bf16 row-major [M x K]; two contiguous 16B loads.
__device__ __forceinline__ v16bf load_a_bf16g(const __bf16* __restrict__ A, int lda,
                                              int m0, int k0, int lane) {
  const int row = m0 + (lane & 15);
  const int kb  = k0 + ((lane >> 4) << 3);
  const __bf16* p = A + (size_t)row * lda + kb;
  v16bf r;
#pragma unroll
  for (int i = 0; i < 8; ++i) r[i] = p[i];
#pragma unroll
  for (int i = 0; i < 8; ++i) r[8 + i] = p[16 + i];
  return r;
}

// B fragment (32x16 bf16) from fp32 row-major [K x N], ldb = N.
// VGPR j: lanes 0-15 hold (K=k0+2j(+8 if j>=4), +1, N=n0+l); lanes 16-31 K += 8.
// Across lanes the 16 addresses per row are consecutive floats -> coalesced.
__device__ __forceinline__ v16bf load_b_f32(const float* __restrict__ B, int ldb,
                                            int k0, int n0, int lane) {
  const int n  = n0 + (lane & 15);
  const int kb = k0 + ((lane >> 4) << 3);
  const float* p = B + (size_t)kb * ldb + n;
  v16bf r;
#pragma unroll
  for (int j = 0; j < 8; ++j) {
    r[2 * j]     = (__bf16)p[(size_t)(2 * j + ((j >> 2) << 3)) * ldb];
    r[2 * j + 1] = (__bf16)p[(size_t)(2 * j + 1 + ((j >> 2) << 3)) * ldb];
  }
  return r;
}

// ---------------------------------------------------------------------------
// Kernel 1: S = relu(F @ Ws + bs), stored bf16 row-major [256 x 512] in d_ws.
// One 16x16 output tile per wave; 512 tiles total.
// ---------------------------------------------------------------------------
__global__ __launch_bounds__(256) void shared_gemm_kernel(
    const float* __restrict__ F, const float* __restrict__ Ws,
    const float* __restrict__ bs, __bf16* __restrict__ S) {
  const int lane = threadIdx.x & 31;
  const int wave = threadIdx.x >> 5;
  const int tile = blockIdx.x * 8 + wave;  // 0..511
  const int m0 = (tile >> 5) << 4;         // 16 M-tiles (B = 256)
  const int n0 = (tile & 31) << 4;         // 32 N-tiles (H = 512)

  v8f acc = vzero8();
  for (int k0 = 0; k0 < 2048; k0 += 32) {
    v16bf a = load_a_f32(F, 2048, m0, k0, lane);
    v16bf b = load_b_f32(Ws, 512, k0, n0, lane);
    acc = wmma_bf16(a, b, acc);
  }

  // C layout: VGPR r -> row m0 + r (+8 for lanes 16-31), col n0 + (lane&15)
  const int col   = n0 + (lane & 15);
  const int rbase = m0 + ((lane >> 4) << 3);
  const float bias = bs[col];
#pragma unroll
  for (int r = 0; r < 8; ++r) {
    float v = acc[r] + bias;
    v = v > 0.0f ? v : 0.0f;
    S[(size_t)(rbase + r) * 512 + col] = (__bf16)v;
  }
}

// ---------------------------------------------------------------------------
// Kernel 2: per class c and B-tile bt: h = relu(S @ W1[c] + b1[c]) (kept in
// LDS as bf16), logits = h @ W2[c] + b2[c], softmax over E=8.
// grid = (2, 512); 256 threads = 8 waves (2 M x 4 N of 64x64 each).
// ---------------------------------------------------------------------------
__global__ __launch_bounds__(256) void gate_kernel(
    const __bf16* __restrict__ S, const float* __restrict__ W1,
    const float* __restrict__ b1, const float* __restrict__ W2,
    const float* __restrict__ b2, float* __restrict__ gw,
    float* __restrict__ gl) {
  __shared__ __bf16 hbuf[128 * 256];  // 64 KB, XOR-swizzled in 8-col groups

  const int lane = threadIdx.x & 31;
  const int wave = threadIdx.x >> 5;
  const int bt = blockIdx.x;  // 0..1   (B tile of 128)
  const int c  = blockIdx.y;  // 0..511 (class)
  const int wm = wave >> 2;   // 0..1
  const int wn = wave & 3;    // 0..3
  const int mg = bt * 128 + wm * 64;  // global row base of this wave
  const int lm = wm * 64;             // local row base within 128-row tile
  const int nb = wn * 64;             // col base of this wave

  const float* W1c = W1 + (size_t)c * (512 * 256);

  // ---- GEMM 1: 64x64 per wave, K = 512 ----
  v8f acc[4][4];
#pragma unroll
  for (int mi = 0; mi < 4; ++mi)
#pragma unroll
    for (int ni = 0; ni < 4; ++ni) acc[mi][ni] = vzero8();

  for (int k0 = 0; k0 < 512; k0 += 32) {
    v16bf a[4], b[4];
#pragma unroll
    for (int mi = 0; mi < 4; ++mi)
      a[mi] = load_a_bf16g(S, 512, mg + 16 * mi, k0, lane);
#pragma unroll
    for (int ni = 0; ni < 4; ++ni)
      b[ni] = load_b_f32(W1c, 256, k0, nb + 16 * ni, lane);
#pragma unroll
    for (int mi = 0; mi < 4; ++mi)
#pragma unroll
      for (int ni = 0; ni < 4; ++ni)
        acc[mi][ni] = wmma_bf16(a[mi], b[ni], acc[mi][ni]);
  }

  // ---- bias + ReLU + bf16 store into swizzled LDS ----
#pragma unroll
  for (int ni = 0; ni < 4; ++ni) {
    const int col = nb + 16 * ni + (lane & 15);
    const float bias = b1[(size_t)c * 256 + col];
#pragma unroll
    for (int mi = 0; mi < 4; ++mi) {
      const int rbase = lm + 16 * mi + ((lane >> 4) << 3);
#pragma unroll
      for (int r = 0; r < 8; ++r) {
        float v = acc[mi][ni][r] + bias;
        v = v > 0.0f ? v : 0.0f;
        const int row = rbase + r;
        const int sc = ((((col >> 3) ^ (row & 15)) << 3) | (col & 7));
        hbuf[row * 256 + sc] = (__bf16)v;
      }
    }
  }
  __syncthreads();

  // ---- GEMM 2: each wave one 16-row tile; N = 16 (cols 8..15 zero), K = 256
  v8f acc2 = vzero8();
  const int  n      = lane & 15;
  const bool nvalid = n < 8;
  const float* W2c  = W2 + (size_t)c * (256 * 8);

  for (int k0 = 0; k0 < 256; k0 += 32) {
    // A fragment from swizzled LDS (two conflict-free 16B ds loads)
    const int row = wave * 16 + (lane & 15);
    const int kb  = k0 + ((lane >> 4) << 3);
    const int g0  = (((kb >> 3) ^ (row & 15)) << 3);
    const int g1  = ((((kb + 16) >> 3) ^ (row & 15)) << 3);
    const __bf16* p0 = &hbuf[row * 256 + g0];
    const __bf16* p1 = &hbuf[row * 256 + g1];
    v16bf a;
#pragma unroll
    for (int i = 0; i < 8; ++i) a[i] = p0[i];
#pragma unroll
    for (int i = 0; i < 8; ++i) a[8 + i] = p1[i];

    // B fragment from W2[c] fp32 [256 x 8]; cols >= 8 padded with zeros
    const int kbb = k0 + ((lane >> 4) << 3);
    v16bf b;
#pragma unroll
    for (int j = 0; j < 8; ++j) {
      float f0 = nvalid ? W2c[(size_t)(kbb + 2 * j + ((j >> 2) << 3)) * 8 + n] : 0.0f;
      float f1 = nvalid ? W2c[(size_t)(kbb + 2 * j + 1 + ((j >> 2) << 3)) * 8 + n] : 0.0f;
      b[2 * j]     = (__bf16)f0;
      b[2 * j + 1] = (__bf16)f1;
    }
    acc2 = wmma_bf16(a, b, acc2);
  }

  // ---- bias + softmax over e=0..7 (butterfly within 8-lane groups) ----
  const float b2v = nvalid ? b2[(size_t)c * 8 + n] : 0.0f;
#pragma unroll
  for (int r = 0; r < 8; ++r) {
    float x = acc2[r] + b2v;
    float m = x;
    m = fmaxf(m, __shfl_xor(m, 1, 32));
    m = fmaxf(m, __shfl_xor(m, 2, 32));
    m = fmaxf(m, __shfl_xor(m, 4, 32));
    float e = __expf(x - m);
    float s = e;
    s += __shfl_xor(s, 1, 32);
    s += __shfl_xor(s, 2, 32);
    s += __shfl_xor(s, 4, 32);
    float w = e / s;
    if (nvalid) {
      const int bg = bt * 128 + wave * 16 + ((lane >> 4) << 3) + r;
      const size_t o = (size_t)bg * (512 * 8) + (size_t)c * 8 + n;
      gw[o] = w;
      gl[o] = x;
    }
  }
}

extern "C" void kernel_launch(void* const* d_in, const int* in_sizes, int n_in,
                              void* d_out, int out_size, void* d_ws, size_t ws_size,
                              hipStream_t stream) {
  const float* F  = (const float*)d_in[0];  // features [256, 2048]
  const float* Ws = (const float*)d_in[1];  // [2048, 512]
  const float* bs = (const float*)d_in[2];  // [512]
  const float* W1 = (const float*)d_in[3];  // [512, 512, 256]
  const float* b1 = (const float*)d_in[4];  // [512, 256]
  const float* W2 = (const float*)d_in[5];  // [512, 256, 8]
  const float* b2 = (const float*)d_in[6];  // [512, 8]

  float* gw = (float*)d_out;                      // gate_weights [256,512,8]
  float* gl = gw + (size_t)256 * 512 * 8;         // gate_logits  [256,512,8]
  __bf16* S = (__bf16*)d_ws;                      // shared acts bf16 [256,512]

  shared_gemm_kernel<<<64, 256, 0, stream>>>(F, Ws, bs, S);
  gate_kernel<<<dim3(2, 512), 256, 0, stream>>>(S, W1, b1, W2, b2, gw, gl);
}
